// IsolaCLIPLoss_28767690949172
// MI455X (gfx1250) — compile-verified, matrix operand order checked by reading, not run on hardware
//
#include <hip/hip_runtime.h>

typedef __attribute__((ext_vector_type(16))) _Float16 v16h;
typedef __attribute__((ext_vector_type(8)))  float    v8f;

#define N_ROWS 8192
#define DIM    512
#define TILE   128                 // 128x128 output tile per workgroup
#define NTILE  (N_ROWS / TILE)     // 64 tile rows
#define KSTEP  32                  // K per WMMA f16 step
#define NSTEPS (DIM / KSTEP)       // 16

union V16 { uint4 u[2]; v16h v; };

// Async copy 16B from global directly into LDS (CDNA5 path, ASYNCcnt-tracked).
// LDS operand = workgroup-relative byte offset (low 32 bits of generic ptr).
__device__ __forceinline__ void async_cp16(const _Float16* g, const uint4* l) {
    unsigned loff = (unsigned)(uintptr_t)l;
    asm volatile("global_load_async_to_lds_b128 %0, %1, off"
                 :: "v"(loff), "v"(g) : "memory");
}

// ---------------------------------------------------------------------------
// Kernel 1: row-wise L2 normalize both embeddings, write f16 copies for WMMA,
// and accumulate align loss partials. One wave per row (512 f32 = 16/lane).
// ---------------------------------------------------------------------------
__global__ void __launch_bounds__(256)
normalize_align_kernel(const float* __restrict__ img,
                       const float* __restrict__ txt,
                       _Float16* __restrict__ himg,
                       _Float16* __restrict__ htxt,
                       float* __restrict__ accum)
{
    const int lane = threadIdx.x & 31;
    const int wave = threadIdx.x >> 5;
    const int row  = blockIdx.x * 8 + wave;
    const size_t base = (size_t)row * DIM;

    float xi[16], xt[16];
    float ssi = 0.f, sst = 0.f;
#pragma unroll
    for (int k = 0; k < 16; ++k) {
        xi[k] = img[base + lane + 32 * k]; ssi += xi[k] * xi[k];
        xt[k] = txt[base + lane + 32 * k]; sst += xt[k] * xt[k];
    }
#pragma unroll
    for (int off = 16; off > 0; off >>= 1) {
        ssi += __shfl_xor(ssi, off, 32);
        sst += __shfl_xor(sst, off, 32);
    }
    const float si = 1.0f / fmaxf(sqrtf(ssi), 1e-12f);
    const float st = 1.0f / fmaxf(sqrtf(sst), 1e-12f);

    float al = 0.f;
#pragma unroll
    for (int k = 0; k < 16; ++k) {
        const float a = xi[k] * si;
        const float b = xt[k] * st;
        const float d = a - b;
        al += d * d;
        himg[base + lane + 32 * k] = (_Float16)a;
        htxt[base + lane + 32 * k] = (_Float16)b;
    }
#pragma unroll
    for (int off = 16; off > 0; off >>= 1) al += __shfl_xor(al, off, 32);
    if (lane == 0) atomicAdd(&accum[0], al);
}

// ---------------------------------------------------------------------------
// Kernel 2: upper-triangular 128x128 Gram tile via v_wmma_f32_16x16x32_f16.
// Double-buffered LDS fed by global_load_async_to_lds_b128 so the next
// K-panel streams in while WMMAs consume the current one. Fused
// exp(min(4G-4,0)) + sum epilogue. grid = (2080, 2): y selects matrix.
// ---------------------------------------------------------------------------
__global__ void __launch_bounds__(256)
gram_exp_sum_kernel(const _Float16* __restrict__ himg,
                    const _Float16* __restrict__ htxt,
                    float* __restrict__ accum)
{
    const _Float16* __restrict__ X = (blockIdx.y == 0) ? himg : htxt;
    float* acc_slot = &accum[1 + blockIdx.y];

    // decode linear block index -> upper-triangular tile pair (bi <= bj)
    int t = blockIdx.x;
    int bi = 0;
    while (t >= NTILE - bi) { t -= (NTILE - bi); ++bi; }
    const int bj = bi + t;
    const float wgt = (bi == bj) ? 1.0f : 2.0f;   // symmetry: off-diag counts twice

    // double-buffered panels: 128 rows x 32 halves, row stride 5 uint4 (80B)
    __shared__ uint4 As[2][TILE][5];
    __shared__ uint4 Bs[2][TILE][5];

    const int tid  = threadIdx.x;
    const int lane = tid & 31;
    const int wave = tid >> 5;
    const int r16  = lane & 15;   // M (A) / N (B) within 16x16 tile
    const int h    = lane >> 4;   // K half-select per ISA 16-bit layout

    const int r0 = tid >> 2;      // staging row (this thread), +64 for 2nd chunk
    const int c0 = tid & 3;       // staging uint4 column

    v8f acc[8];
#pragma unroll
    for (int j = 0; j < 8; ++j) acc[j] = (v8f){};

    const size_t rowA = (size_t)bi * TILE;
    const size_t rowB = (size_t)bj * TILE;

    // prologue: stage K-step 0 into buffer 0 (4 async ops per thread/wave)
    async_cp16(X + (rowA + r0)      * DIM + c0 * 8, &As[0][r0][c0]);
    async_cp16(X + (rowA + r0 + 64) * DIM + c0 * 8, &As[0][r0 + 64][c0]);
    async_cp16(X + (rowB + r0)      * DIM + c0 * 8, &Bs[0][r0][c0]);
    async_cp16(X + (rowB + r0 + 64) * DIM + c0 * 8, &Bs[0][r0 + 64][c0]);

    for (int step = 0; step < NSTEPS; ++step) {
        const int buf = step & 1;
        // retire reads of buf^1 (last touched in step-1) before refilling it
        __syncthreads();
        if (step + 1 < NSTEPS) {
            const int kk = (step + 1) * KSTEP;
            async_cp16(X + (rowA + r0)      * DIM + kk + c0 * 8, &As[buf ^ 1][r0][c0]);
            async_cp16(X + (rowA + r0 + 64) * DIM + kk + c0 * 8, &As[buf ^ 1][r0 + 64][c0]);
            async_cp16(X + (rowB + r0)      * DIM + kk + c0 * 8, &Bs[buf ^ 1][r0][c0]);
            async_cp16(X + (rowB + r0 + 64) * DIM + kk + c0 * 8, &Bs[buf ^ 1][r0 + 64][c0]);
            // in-order per wave: <=4 outstanding means current buffer landed
            asm volatile("s_wait_asynccnt 0x4" ::: "memory");
        } else {
            asm volatile("s_wait_asynccnt 0x0" ::: "memory");
        }
        __syncthreads();   // current buffer visible to all waves

        // A fragment: lane holds K = {h*8..h*8+7} and {16+h*8..16+h*8+7}
        V16 a;
        a.u[0] = As[buf][wave * 16 + r16][h];
        a.u[1] = As[buf][wave * 16 + r16][2 + h];
#pragma unroll
        for (int j = 0; j < 8; ++j) {
            V16 b;   // B = X^T: column n of B is row (bj*128 + j*16 + n) of X
            b.u[0] = Bs[buf][j * 16 + r16][h];
            b.u[1] = Bs[buf][j * 16 + r16][2 + h];
            acc[j] = __builtin_amdgcn_wmma_f32_16x16x32_f16(
                false, a.v, false, b.v, (short)0, acc[j], false, false);
        }
    }

    // epilogue: d2 = max(2-2G,0)  ->  exp(-2*d2) = exp(min(4G-4, 0))
    float p = 0.f;
#pragma unroll
    for (int j = 0; j < 8; ++j)
#pragma unroll
        for (int v = 0; v < 8; ++v)
            p += __expf(fminf(4.0f * acc[j][v] - 4.0f, 0.0f));
    p *= wgt;
#pragma unroll
    for (int off = 16; off > 0; off >>= 1) p += __shfl_down(p, off, 32);
    if (lane == 0) atomicAdd(acc_slot, p);
}

// ---------------------------------------------------------------------------
// Kernel 3: combine scalars.
// ---------------------------------------------------------------------------
__global__ void finalize_kernel(const float* __restrict__ accum,
                                float* __restrict__ out)
{
    if (threadIdx.x == 0 && blockIdx.x == 0) {
        const float nn     = (float)N_ROWS * (float)N_ROWS;
        const float align  = accum[0] / (float)N_ROWS;
        const float s_img  = accum[1];
        const float s_txt  = accum[2];
        const float unif   = 0.5f * (logf(s_txt / nn) + logf(s_img / nn));
        out[0] = 3.0f * align + 1.0f * unif;   // ALIGN_SCALE=3, UNIF_SCALE=1
    }
}

extern "C" void kernel_launch(void* const* d_in, const int* in_sizes, int n_in,
                              void* d_out, int out_size, void* d_ws, size_t ws_size,
                              hipStream_t stream) {
    const float* img = (const float*)d_in[0];
    const float* txt = (const float*)d_in[1];
    float* out = (float*)d_out;

    // workspace: f16 normalized copies (8 MB each) + 3 float accumulators
    _Float16* himg = (_Float16*)d_ws;
    _Float16* htxt = himg + (size_t)N_ROWS * DIM;
    float* accum   = (float*)(htxt + (size_t)N_ROWS * DIM);

    hipMemsetAsync(accum, 0, 3 * sizeof(float), stream);

    normalize_align_kernel<<<N_ROWS / 8, 256, 0, stream>>>(img, txt, himg, htxt, accum);

    dim3 grid(NTILE * (NTILE + 1) / 2, 2);   // (2080, 2)
    gram_exp_sum_kernel<<<grid, 256, 0, stream>>>(himg, htxt, accum);

    finalize_kernel<<<1, 32, 0, stream>>>(accum, out);
}
